// OMNIANOMALY_60112362275054
// MI455X (gfx1250) — matrix operand order repaired and checked by reading, not compiled, not run on hardware
//
#include <hip/hip_runtime.h>
#include <hip/hip_bf16.h>

// ---------------------------------------------------------------------------
// OmniAnomaly forward, MI455X (gfx1250), wave32 + WMMA bf16.
// B=128, T=64, X=Z=64, H=D=512, 3H=1536, K_flow=2.
// One persistent workgroup per 16 batch rows (8 WGs) runs the whole
// encoder+decoder recurrence with no inter-WG sync (row-independent GRU).
// Weight fragments stream from the L2-resident bf16 weight copies every
// timestep. Pointer "laundering" stops LICM from hoisting fragments out of
// the time loop (scratch spills), and explicit addrspace(1) accesses keep
// the streams on global_load_b128 (not flat_load, which also ties up DScnt
// and forces 64-bit VGPR addressing).
// ---------------------------------------------------------------------------

typedef __bf16 bf16_t;
typedef bf16_t v16bf __attribute__((ext_vector_type(16)));
typedef float  v8f   __attribute__((ext_vector_type(8)));
typedef unsigned int v4u __attribute__((ext_vector_type(4)));   // POD 16-byte vector
typedef unsigned short u16;

union Frag { v16bf v; v4u q[2]; };

// Defeat loop-invariant code motion: make p look redefined each iteration.
#define LAUNDER(p) asm volatile("" : "+s"(p))

// Force the GLOBAL (addrspace 1) path: global_load_b128 / global_store_b32.
typedef v4u   __attribute__((address_space(1))) g_v4u;
typedef u16   __attribute__((address_space(1))) g_u16;
typedef float __attribute__((address_space(1))) g_f32;

__device__ __forceinline__ v4u gload16(const void* p) {
    return *(const g_v4u*)(unsigned long long)p;
}
__device__ __forceinline__ u16 gload_u16(const void* p) {
    return *(const g_u16*)(unsigned long long)p;
}
__device__ __forceinline__ float gload_f32(const void* p) {
    return *(const g_f32*)(unsigned long long)p;
}
__device__ __forceinline__ void gstore_f32(void* p, float v) {
    *(g_f32*)(unsigned long long)p = v;
}

__device__ __forceinline__ u16 f2bf(float f) {
    union { float f; unsigned u; } c; c.f = f;
    unsigned u = c.u;
    return (u16)((u + 0x7FFFu + ((u >> 16) & 1u)) >> 16);
}

__device__ __forceinline__ v8f vzero8() {
    v8f z;
#pragma unroll
    for (int i = 0; i < 8; ++i) z[i] = 0.f;
    return z;
}

// A-fragment (16x32 bf16, M x K) from LDS: lane<16 holds row=lane,
// K = k0..k0+15 contiguous; lane>=16 holds row=lane-16, K = k0+16..31.
__device__ __forceinline__ void load_a(const u16* A, int lda, int k0, int lane, Frag& f) {
    int half = lane >> 4, row = lane & 15;
    const u16* p = A + row * lda + k0 + half * 16;
    f.q[0] = *(const v4u*)(p);
    f.q[1] = *(const v4u*)(p + 8);
}

// B-fragment (32x16 bf16, K x N) for out = in @ W^T, W row-major [N_out, K]
// in GLOBAL memory: lane n<16: col j0+n, K=k0..k0+15 contiguous;
// lane>=16: col j0+n-16, K=k0+16..31.
__device__ __forceinline__ void load_b(const u16* W, int ldw, int j0, int k0, int lane, Frag& f) {
    int half = lane >> 4, n = lane & 15;
    const u16* p = W + (size_t)(j0 + n) * ldw + k0 + half * 16;
    f.q[0] = gload16(p);
    f.q[1] = gload16(p + 8);
}

__device__ __forceinline__ v8f wmma_bf(const Frag& a, const Frag& b, v8f c) {
    return __builtin_amdgcn_wmma_f32_16x16x32_bf16(false, a.v, false, b.v,
                                                   (short)0, c, false, false);
}

// acc += A[16 x K] @ W[j0:j0+16, 0:K]^T   (A in LDS, W in global/L2)
__device__ __forceinline__ v8f gemm_tile(const u16* A, int lda,
                                         const u16* W, int ldw,
                                         int j0, int K, int lane, v8f acc) {
    for (int k0 = 0; k0 < K; k0 += 32) {
        Frag a, b;
        load_a(A, lda, k0, lane, a);
        load_b(W, ldw, j0, k0, lane, b);
        acc = wmma_bf(a, b, acc);
    }
    return acc;
}

__device__ __forceinline__ float sigm(float x)  { return 1.f / (1.f + __expf(-x)); }
__device__ __forceinline__ float softp(float x) { return x > 20.f ? x : log1pf(__expf(x)); }

__global__ void cvt_bf16_kernel(const float* __restrict__ src, u16* __restrict__ dst, int n) {
    int i = blockIdx.x * blockDim.x + threadIdx.x;
    if (i < n) dst[i] = f2bf(src[i]);
}

#define TB 64   // timesteps
#define HB 512
#define ZB 64
#define BB 128

__global__ __launch_bounds__(512)
void omni_main_kernel(
    const float* __restrict__ x, const float* __restrict__ eps_enc,
    const float* __restrict__ eps_dec,
    const float* __restrict__ bi_enc, const float* __restrict__ bh_enc,
    const float* __restrict__ b_phi_enc,
    const float* __restrict__ b_zmean, const float* __restrict__ b_zstd,
    const float* __restrict__ w_flow, const float* __restrict__ b_flow,
    const float* __restrict__ u_flow,
    const float* __restrict__ bi_dec, const float* __restrict__ bh_dec,
    const float* __restrict__ b_phi_dec,
    const float* __restrict__ b_xmean, const float* __restrict__ b_xstd,
    const u16* __restrict__ wWiE, const u16* __restrict__ wWhE,
    const u16* __restrict__ wWpE, const u16* __restrict__ wWzm,
    const u16* __restrict__ wWzs,
    const u16* __restrict__ wWiD, const u16* __restrict__ wWhD,
    const u16* __restrict__ wWpD, const u16* __restrict__ wWxm,
    const u16* __restrict__ wWxs,
    const u16* __restrict__ xbf,
    float* __restrict__ out, float* __restrict__ partials)
{
    const int tid  = threadIdx.x;
    const int wave = tid >> 5;
    const int lane = tid & 31;
    const int half = lane >> 4;
    const int nn   = lane & 15;
    const int b0   = blockIdx.x * 16;

    __shared__ u16  hbf[16 * 520];      // h (bf16) for WMMA A
    __shared__ u16  phibf[16 * 520];    // phi (bf16)
    __shared__ u16  inbf[16 * 80];      // per-step input tile (x_t or z_t)
    __shared__ float m_s[16 * 64];      // mean head
    __shared__ float s_s[16 * 64];      // std head
    __shared__ float zb[16 * 64];       // z being flowed
    __shared__ float bias_ru[1024];
    __shared__ float bias_in[512];
    __shared__ float bias_hn[512];
    __shared__ float bias_phi[512];
    __shared__ float bias_m[64];
    __shared__ float bias_sd[64];
    __shared__ float sflow[16];
    __shared__ float redn[16], redk[16];

    const size_t SEG = (size_t)BB * TB * ZB;   // 524288
    float* OUT_Z  = out + 2;
    float* OUT_ZM = OUT_Z + SEG;
    float* OUT_ZS = OUT_Z + 2 * SEG;
    float* OUT_XO = OUT_Z + 3 * SEG;
    float* OUT_XM = OUT_Z + 4 * SEG;
    float* OUT_XS = OUT_Z + 5 * SEG;

    float nll_acc = 0.f, kld_acc = 0.f;
    float hreg[2][8];   // f32 master hidden state, WMMA C layout

    for (int phase = 0; phase < 2; ++phase) {
        const u16* Wi   = phase ? wWiD : wWiE;
        const u16* Wh   = phase ? wWhD : wWhE;
        const u16* Wp   = phase ? wWpD : wWpE;
        const int  ldp  = phase ? 512 : 576;   // W_phi_enc is [512, H+Z]
        const u16* Wm   = phase ? wWxm : wWzm;
        const u16* Ws   = phase ? wWxs : wWzs;
        const float* biF  = phase ? bi_dec : bi_enc;
        const float* bhF  = phase ? bh_dec : bh_enc;
        const float* bpF  = phase ? b_phi_dec : b_phi_enc;
        const float* bmF  = phase ? b_xmean : b_zmean;
        const float* bsF  = phase ? b_xstd  : b_zstd;
        float* OUT_M = phase ? OUT_XM : OUT_ZM;
        float* OUT_S = phase ? OUT_XS : OUT_ZS;

        for (int i = tid; i < 1024; i += 512) bias_ru[i] = biF[i] + bhF[i];
        if (tid < 512) {
            bias_in[tid]  = biF[1024 + tid];
            bias_hn[tid]  = bhF[1024 + tid];
            bias_phi[tid] = bpF[tid];
        }
        if (tid < 64) { bias_m[tid] = bmF[tid]; bias_sd[tid] = bsF[tid]; }
        for (int i = tid; i < 16 * 520; i += 512) hbf[i] = 0;
#pragma unroll
        for (int r = 0; r < 8; ++r) { hreg[0][r] = 0.f; hreg[1][r] = 0.f; }
        if (phase) __threadfence();   // make encoder's OUT_Z writes visible
        __syncthreads();

        for (int t = 0; t < TB; ++t) {
            // Pin weight streams inside the time loop: without this, LICM
            // hoists hundreds of B-fragments out of the t-loop and spills
            // them to scratch.
            LAUNDER(Wi); LAUNDER(Wh); LAUNDER(Wp); LAUNDER(Wm); LAUNDER(Ws);

            // ---- stage input tile (bf16) into LDS ----
            for (int e = tid; e < 1024; e += 512) {
                int row = e >> 6, c = e & 63;
                if (phase == 0)
                    inbf[row * 80 + c] = gload_u16(xbf + ((size_t)(b0 + row) * TB + t) * 64 + c);
                else
                    inbf[row * 80 + c] = f2bf(gload_f32(OUT_Z + ((size_t)(b0 + row) * TB + t) * 64 + c));
            }
            __syncthreads();

            // ---- GRU gates: gi + gh via WMMA (2 chunks of 16 cols per wave) ----
            v8f aR[2], aU[2], aNi[2], aNh[2];
#pragma unroll
            for (int c = 0; c < 2; ++c) {
                int j0 = (wave + 16 * c) * 16;
                aR[c]  = gemm_tile(inbf, 80, Wi, 64, j0,        64,  lane, vzero8());
                aR[c]  = gemm_tile(hbf, 520, Wh, 512, j0,       512, lane, aR[c]);
                aU[c]  = gemm_tile(inbf, 80, Wi, 64, j0 + 512,  64,  lane, vzero8());
                aU[c]  = gemm_tile(hbf, 520, Wh, 512, j0 + 512, 512, lane, aU[c]);
                aNi[c] = gemm_tile(inbf, 80, Wi, 64, j0 + 1024, 64,  lane, vzero8());
                aNh[c] = gemm_tile(hbf, 520, Wh, 512, j0 + 1024, 512, lane, vzero8());
            }
            __syncthreads();   // everyone done reading hbf

            // ---- GRU elementwise on accumulator registers ----
#pragma unroll
            for (int c = 0; c < 2; ++c) {
                int j0 = (wave + 16 * c) * 16;
                int col = j0 + nn;
                float br = bias_ru[col], bu = bias_ru[512 + col];
                float bnI = bias_in[col], bnH = bias_hn[col];
#pragma unroll
                for (int r = 0; r < 8; ++r) {
                    float rv = sigm(aR[c][r] + br);
                    float uv = sigm(aU[c][r] + bu);
                    float nv = tanhf(aNi[c][r] + bnI + rv * (aNh[c][r] + bnH));
                    float hn = (1.f - uv) * nv + uv * hreg[c][r];
                    hreg[c][r] = hn;
                    hbf[(r + 8 * half) * 520 + col] = f2bf(hn);
                }
            }
            __syncthreads();

            // ---- phi = relu(h @ Wphi[:, :H]^T + b) ----
#pragma unroll
            for (int c = 0; c < 2; ++c) {
                int j0 = (wave + 16 * c) * 16;
                v8f p = gemm_tile(hbf, 520, Wp, ldp, j0, 512, lane, vzero8());
                int col = j0 + nn;
                float bp = bias_phi[col];
#pragma unroll
                for (int r = 0; r < 8; ++r) {
                    float v = p[r] + bp;
                    v = v > 0.f ? v : 0.f;
                    phibf[(r + 8 * half) * 520 + col] = f2bf(v);
                }
            }
            __syncthreads();

            // ---- heads: mean (sigmoid) / std (softplus), 8 tiles on waves 0..7 ----
            if (wave < 8) {
                int hs = wave >> 2;                 // 0=mean, 1=std
                int j0 = (wave & 3) * 16;
                const u16* W = hs ? Ws : Wm;
                v8f ha = gemm_tile(phibf, 520, W, 512, j0, 512, lane, vzero8());
                int col = j0 + nn;
                float bb = hs ? bias_sd[col] : bias_m[col];
#pragma unroll
                for (int r = 0; r < 8; ++r) {
                    float xv = ha[r] + bb;
                    float val = hs ? softp(xv) : sigm(xv);
                    int M = r + 8 * half;
                    (hs ? s_s : m_s)[M * 64 + col] = val;
                    gstore_f32((hs ? OUT_S : OUT_M) + ((size_t)(b0 + M) * TB + t) * 64 + col, val);
                }
            }
            __syncthreads();

            if (phase == 0) {
                // z = eps * std + mean ; KLD(z) partial
                for (int e = tid; e < 1024; e += 512) {
                    int row = e >> 6, c = e & 63;
                    float zmv = m_s[e], zsv = s_s[e];
                    float ep = gload_f32(eps_enc + ((size_t)(b0 + row) * TB + t) * 64 + c);
                    zb[e] = ep * zsv + zmv;
                    kld_acc += 1.f + zsv - zmv * zmv - __expf(zsv);
                }
                __syncthreads();
                // planar flows (K=2, sequential)
                for (int k = 0; k < 2; ++k) {
                    if (tid < 16) {
                        float s = gload_f32(b_flow + t * 2 + k);
                        const float* wf = w_flow + (t * 2 + k) * 64;
                        for (int z = 0; z < 64; ++z) s += zb[tid * 64 + z] * gload_f32(wf + z);
                        sflow[tid] = tanhf(s);
                    }
                    __syncthreads();
                    for (int e = tid; e < 1024; e += 512) {
                        int row = e >> 6, c = e & 63;
                        zb[e] += gload_f32(u_flow + (t * 2 + k) * 64 + c) * sflow[row];
                    }
                    __syncthreads();
                }
                for (int e = tid; e < 1024; e += 512) {
                    int row = e >> 6, c = e & 63;
                    gstore_f32(OUT_Z + ((size_t)(b0 + row) * TB + t) * 64 + c, zb[e]);
                }
            } else {
                // xo = eps * std + mean ; NLL + KLD(x) partials
                for (int e = tid; e < 1024; e += 512) {
                    int row = e >> 6, c = e & 63;
                    size_t idx = ((size_t)(b0 + row) * TB + t) * 64 + c;
                    float xmv = m_s[e], xsv = s_s[e];
                    gstore_f32(OUT_XO + idx, gload_f32(eps_dec + idx) * xsv + xmv);
                    float xv = gload_f32(x + idx);
                    float d = (xv - xmv) * __expf(-0.5f * xsv);
                    nll_acc += 0.5f * (xsv + d * d);
                    kld_acc += 1.f + xsv - xmv * xmv - __expf(xsv);
                }
            }
            __syncthreads();
        }
        __syncthreads();
    }

    // ---- block reduction of loss partials (wave32 shuffles) ----
    for (int off = 16; off > 0; off >>= 1) {
        nll_acc += __shfl_down(nll_acc, off);
        kld_acc += __shfl_down(kld_acc, off);
    }
    if (lane == 0) { redn[wave] = nll_acc; redk[wave] = kld_acc; }
    __syncthreads();
    if (tid == 0) {
        float n = 0.f, k = 0.f;
        for (int i = 0; i < 16; ++i) { n += redn[i]; k += redk[i]; }
        partials[blockIdx.x * 2 + 0] = n;
        partials[blockIdx.x * 2 + 1] = k;
    }
}

__global__ void omni_finalize_kernel(const float* __restrict__ partials, float* __restrict__ out) {
    if (threadIdx.x == 0) {
        float n = 0.f, k = 0.f;
        for (int i = 0; i < 8; ++i) { n += partials[2 * i]; k += partials[2 * i + 1]; }
        out[0] = n;
        out[1] = -0.5f * k;
    }
}

extern "C" void kernel_launch(void* const* d_in, const int* in_sizes, int n_in,
                              void* d_out, int out_size, void* d_ws, size_t ws_size,
                              hipStream_t stream) {
    const float* x        = (const float*)d_in[0];
    const float* eps_enc  = (const float*)d_in[1];
    const float* eps_dec  = (const float*)d_in[2];
    const float* Wi_enc   = (const float*)d_in[3];
    const float* Wh_enc   = (const float*)d_in[4];
    const float* bi_enc   = (const float*)d_in[5];
    const float* bh_enc   = (const float*)d_in[6];
    const float* Wphi_enc = (const float*)d_in[7];
    const float* bphi_enc = (const float*)d_in[8];
    const float* Wzmean   = (const float*)d_in[9];
    const float* bzmean   = (const float*)d_in[10];
    const float* Wzstd    = (const float*)d_in[11];
    const float* bzstd    = (const float*)d_in[12];
    const float* w_flow   = (const float*)d_in[13];
    const float* b_flow   = (const float*)d_in[14];
    const float* u_flow   = (const float*)d_in[15];
    const float* Wi_dec   = (const float*)d_in[16];
    const float* Wh_dec   = (const float*)d_in[17];
    const float* bi_dec   = (const float*)d_in[18];
    const float* bh_dec   = (const float*)d_in[19];
    const float* Wphi_dec = (const float*)d_in[20];
    const float* bphi_dec = (const float*)d_in[21];
    const float* Wxmean   = (const float*)d_in[22];
    const float* bxmean   = (const float*)d_in[23];
    const float* Wxstd    = (const float*)d_in[24];
    const float* bxstd    = (const float*)d_in[25];
    float* out = (float*)d_out;

    // workspace carve (bf16 weight copies + bf16 x + loss partials)
    u16* W = (u16*)d_ws;
    size_t o = 0;
    auto alloc = [&](size_t n) { u16* p = W + o; o += n; return p; };
    u16* wsWiE = alloc(1536 * 64);
    u16* wsWhE = alloc(1536 * 512);
    u16* wsWpE = alloc(512 * 576);
    u16* wsWzm = alloc(64 * 512);
    u16* wsWzs = alloc(64 * 512);
    u16* wsWiD = alloc(1536 * 64);
    u16* wsWhD = alloc(1536 * 512);
    u16* wsWpD = alloc(512 * 512);
    u16* wsWxm = alloc(64 * 512);
    u16* wsWxs = alloc(64 * 512);
    u16* wsX   = alloc((size_t)128 * 64 * 64);
    o = (o + 7) & ~(size_t)7;
    float* partials = (float*)(W + o);

    auto cvt = [&](const float* s, u16* d, int n) {
        cvt_bf16_kernel<<<(n + 255) / 256, 256, 0, stream>>>(s, d, n);
    };
    cvt(Wi_enc, wsWiE, 1536 * 64);
    cvt(Wh_enc, wsWhE, 1536 * 512);
    cvt(Wphi_enc, wsWpE, 512 * 576);
    cvt(Wzmean, wsWzm, 64 * 512);
    cvt(Wzstd, wsWzs, 64 * 512);
    cvt(Wi_dec, wsWiD, 1536 * 64);
    cvt(Wh_dec, wsWhD, 1536 * 512);
    cvt(Wphi_dec, wsWpD, 512 * 512);
    cvt(Wxmean, wsWxm, 64 * 512);
    cvt(Wxstd, wsWxs, 64 * 512);
    cvt(x, wsX, 128 * 64 * 64);

    omni_main_kernel<<<8, 512, 0, stream>>>(
        x, eps_enc, eps_dec,
        bi_enc, bh_enc, bphi_enc, bzmean, bzstd,
        w_flow, b_flow, u_flow,
        bi_dec, bh_dec, bphi_dec, bxmean, bxstd,
        wsWiE, wsWhE, wsWpE, wsWzm, wsWzs,
        wsWiD, wsWhD, wsWpD, wsWxm, wsWxs,
        wsX, out, partials);

    omni_finalize_kernel<<<1, 32, 0, stream>>>(partials, out);
}
